// Block_17025250361620
// MI455X (gfx1250) — matrix operand order repaired and checked by reading, not compiled, not run on hardware
//
#include <hip/hip_runtime.h>
#include <math.h>

// ---------------------------------------------------------------------------
// Transformer block (DyT + RoPE attention + top-2 MoE + shared expert) for
// MI455X (gfx1250). All matmuls run on v_wmma_f32_16x16x32_f16 (wave32).
// Dense GEMMs stage their B tile-column into LDS via the Tensor Data Mover
// (tensor_load_to_lds + s_wait_tensorcnt), shared by all 4 waves of a block.
//
// Shapes: B=1, S=512, D=512, H=8, HD=64, E=16, topk=2, FF=512, ES=1.
// ---------------------------------------------------------------------------

typedef __attribute__((ext_vector_type(16))) _Float16 v16h;
typedef __attribute__((ext_vector_type(8)))  float    v8f;
typedef __attribute__((ext_vector_type(4)))  unsigned v4u;
typedef __attribute__((ext_vector_type(4)))  int      v4i;
typedef __attribute__((ext_vector_type(8)))  int      v8i;

constexpr int cS  = 512;   // sequence
constexpr int cD  = 512;   // d_model
constexpr int cH  = 8;     // heads
constexpr int cHD = 64;    // head dim
constexpr int cE  = 16;    // experts
constexpr int cTK = 2;     // top-k
constexpr int cFF = 512;   // expert hidden
constexpr int MAX_TILES = 80;   // sum ceil(cnt_e/16) <= 1024/16 + 16 = 80

#if __has_builtin(__builtin_amdgcn_tensor_load_to_lds) && \
    __has_builtin(__builtin_amdgcn_s_wait_tensorcnt)
#define USE_TDM 1
#else
#define USE_TDM 0
#endif

// --------------------------- fragment index helpers ------------------------
// 16-bit A fragment (16x32, MxK), ISA 7.12.2:
//   lane L (0..31), half h (0..15):  m = L&15,  k = (h&7) + ((h>>3)<<4) + ((L>>4)<<3)
// 16-bit B fragment (32x16, KxN):
//   lane L, half h:                  n = L&15,  k = (h&7) + ((h>>3)<<4) + ((L>>4)<<3)
// 32-bit C/D (16x16): vgpr r, lane L: m = r + (L>>4)*8, n = L&15
__device__ __forceinline__ int afrag_k(int lane, int h) {
  return (h & 7) + ((h >> 3) << 4) + ((lane >> 4) << 3);
}

__device__ __forceinline__ v8f wmma_f16(v16h a, v16h b, v8f c) {
  return __builtin_amdgcn_wmma_f32_16x16x32_f16(false, a, false, b, (short)0, c,
                                                false, false);
}

__device__ __forceinline__ float gelu_tanh(float x) {
  float x3 = x * x * x;
  return 0.5f * x * (1.0f + tanhf(0.79788456f * (x + 0.044715f * x3)));
}

// half-wave (16 lane) butterfly reductions; xor masks < 16 stay in-group
__device__ __forceinline__ float hmax16(float v) {
  for (int m = 1; m < 16; m <<= 1) v = fmaxf(v, __shfl_xor(v, m, 32));
  return v;
}
__device__ __forceinline__ float hsum16(float v) {
  for (int m = 1; m < 16; m <<= 1) v += __shfl_xor(v, m, 32);
  return v;
}

// scatter-store one C element (mloc, kk) into A-fragment-layout f16 buffer
__device__ __forceinline__ void store_apack(_Float16* dst, int rowTile, int KTdst,
                                            int mloc, int kk, float v) {
  int kt  = kk >> 5;
  int kin = kk & 31;
  int hiK = (kin >> 3) & 1;
  int h   = (kin & 7) + ((kin >> 4) << 3);
  int ln  = mloc + (hiK << 4);
  dst[((size_t)(rowTile * KTdst + kt) * 32 + ln) * 16 + h] = (_Float16)v;
}

#if USE_TDM
// --------------------- Tensor Data Mover 1-D copy helper -------------------
// D# group0: [1:0]=count=1, [63:32]=lds_addr, [120:64]=global_addr,
//            [127:126]=type=2. Group1: data_size=8B, tensor/tile dim0 = n8,
//            tensor_dim1=1, tile_dim1=1, dim0_stride=n8. Groups 2/3 unused.
__device__ __forceinline__ void tdm_load_1d(unsigned ldsOff, const void* gsrc,
                                            unsigned bytes) {
  unsigned n8 = bytes >> 3;                       // 8-byte elements
  unsigned long long ga = (unsigned long long)gsrc;
  v4u g0;
  g0[0] = 1u;                                     // count=1 (user descriptor)
  g0[1] = ldsOff;                                 // LDS byte address
  g0[2] = (unsigned)ga;                           // global addr [31:0]
  g0[3] = (unsigned)((ga >> 32) & 0x01FFFFFFu) | (2u << 30);  // [56:32] | type=2
  v8i g1;
  g1[0] = (int)(3u << 16);                        // data_size = 3 (8 bytes)
  g1[1] = (int)((n8 & 0xFFFFu) << 16);            // tensor_dim0[15:0] @ bits[63:48]
  g1[2] = (int)(((n8 >> 16) & 0xFFFFu) | (1u << 16)); // tensor_dim0[31:16], tensor_dim1=1
  g1[3] = (int)((n8 & 0xFFFFu) << 16);            // tile_dim0 @ bits[127:112]
  g1[4] = 1;                                      // tile_dim1 = 1
  g1[5] = (int)n8;                                // tensor_dim0_stride[31:0]
  g1[6] = 0;
  g1[7] = 0;
  v4i z4 = {0, 0, 0, 0};
#if __clang_major__ >= 23
  v8i z8 = {0, 0, 0, 0, 0, 0, 0, 0};
  __builtin_amdgcn_tensor_load_to_lds(g0, g1, z4, z4, z8, 0);
#else
  __builtin_amdgcn_tensor_load_to_lds(g0, g1, z4, z4, 0);
#endif
}
#endif

// ------------------------------- utility kernels ---------------------------
__global__ void zero_kernel(float* p, int n) {
  int i = blockIdx.x * blockDim.x + threadIdx.x;
  if (i < n) p[i] = 0.0f;
}

__global__ void dyt_kernel(const float* __restrict__ x, const float* __restrict__ a,
                           const float* __restrict__ g, const float* __restrict__ b,
                           float* __restrict__ o, int n) {
  int i = blockIdx.x * blockDim.x + threadIdx.x;
  if (i >= n) return;
  int d = i & (cD - 1);
  o[i] = g[d] * tanhf(a[0] * x[i]) + b[d];
}

__global__ void final_add_kernel(const float* __restrict__ x1, const float* __restrict__ moe,
                                 const float* __restrict__ sh, float* __restrict__ out, int n) {
  int i = blockIdx.x * blockDim.x + threadIdx.x;
  if (i < n) out[i] = x1[i] + moe[i] + sh[i];
}

__global__ void rope_kernel(float* __restrict__ q, float* __restrict__ k) {
  int idx = blockIdx.x * blockDim.x + threadIdx.x;   // S*H*32 total
  if (idx >= cS * cH * 32) return;
  int i  = idx & 31;
  int hh = (idx >> 5) & (cH - 1);
  int s  = idx >> 8;
  float inv = __powf(10000.0f, -(float)i / 32.0f);
  float ang = (float)s * inv;
  float c = __cosf(ang), sn = __sinf(ang);
  size_t base = (size_t)s * cD + hh * cHD + i;
  float q1 = q[base], q2 = q[base + 32];
  q[base]      = q1 * c - q2 * sn;
  q[base + 32] = q1 * sn + q2 * c;
  float k1 = k[base], k2 = k[base + 32];
  k[base]      = k1 * c - k2 * sn;
  k[base + 32] = k1 * sn + k2 * c;
}

// ------------------------------- pack kernels ------------------------------
// A-layout pack: grid.x = MT*KT, grid.z = matrix index, block = 32
__global__ void pack_a(const float* __restrict__ src, _Float16* __restrict__ dst,
                       int KT, int ld, int colOff, int colStep,
                       size_t srcMatStride, size_t dstMatStride) {
  int frag = blockIdx.x;
  int mt = frag / KT, kt = frag % KT;
  int lane = threadIdx.x;
  const float* s = src + (size_t)blockIdx.z * srcMatStride +
                   colOff + blockIdx.z * colStep;
  _Float16* d = dst + (size_t)blockIdx.z * dstMatStride + ((size_t)frag * 32 + lane) * 16;
  int m = mt * 16 + (lane & 15);
#pragma unroll
  for (int h = 0; h < 16; ++h) {
    int k = afrag_k(lane, h);
    d[h] = (_Float16)s[(size_t)m * ld + kt * 32 + k];
  }
}

// B-layout pack: element (k,n) = src[k*kStride + n*nStride + off]
// grid.x = NT*KT, grid.z = matrix index, block = 32
__global__ void pack_b(const float* __restrict__ src, _Float16* __restrict__ dst,
                       int KT, int kStride, int nStride, int colOff, int colStep,
                       size_t srcMatStride, size_t dstMatStride) {
  int frag = blockIdx.x;
  int nt = frag / KT, kt = frag % KT;
  int lane = threadIdx.x;
  size_t off = (size_t)blockIdx.z * srcMatStride + colOff + blockIdx.z * colStep;
  _Float16* d = dst + (size_t)blockIdx.z * dstMatStride + ((size_t)frag * 32 + lane) * 16;
  int n = nt * 16 + (lane & 15);
  int kbase = kt * 32 + ((lane >> 4) << 3);
#pragma unroll
  for (int h = 0; h < 16; ++h)
    d[h] = (_Float16)src[(size_t)(kbase + (h & 7) + ((h >> 3) << 4)) * kStride + (size_t)n * nStride + off];
}

// ------------------------------- dense GEMM --------------------------------
// C[M x N] = act(alpha * A*B + bias) (+resid), optional f16 A-layout store.
// block = 128 (4 waves), grid = (MT/4, NT). Per-wave 16x16 tile, K step 32.
// B tile-column (KT*1024 bytes, contiguous in pack layout) is DMA'd to LDS by
// the TDM once per block and shared by all 4 waves.
__global__ void gemm16(const _Float16* __restrict__ Ap, const _Float16* __restrict__ Bp,
                       float* __restrict__ Cout, const float* __restrict__ resid,
                       const float* __restrict__ biasN, _Float16* __restrict__ apackOut,
                       int KT, int N, int KTdst, int actGelu, float alpha) {
#if USE_TDM
  __shared__ _Float16 bstage[16 * 512];            // up to KT=16 -> 16 KB
#endif
  int lane = threadIdx.x & 31;
  int mt = blockIdx.x * (blockDim.x >> 5) + (threadIdx.x >> 5);
  int nt = blockIdx.y;
  const _Float16* a = Ap + (size_t)mt * KT * 512;   // KT*32*16 halves per tile row
  const _Float16* b = Bp + (size_t)nt * KT * 512;
#if USE_TDM
  if (KT > 0x0FFFFFFF) bstage[0] = (_Float16)0.0f;  // never taken: keep LDS "written"
  if (threadIdx.x == 0) {
    tdm_load_1d(0u, b, (unsigned)KT * 1024u);       // DMA B column -> LDS
    __builtin_amdgcn_s_wait_tensorcnt(0);           // TENSORcnt == 0
  }
  __syncthreads();                                  // publish LDS to all 4 waves
#endif
  v8f acc = {};
  for (int kt = 0; kt < KT; ++kt) {
    v16h af = *(const v16h*)(a + ((size_t)kt * 32 + lane) * 16);
    if (kt + 1 < KT)   // gfx1250 global_prefetch_b8 for the next A fragment
      __builtin_prefetch(a + ((size_t)(kt + 1) * 32 + lane) * 16, 0, 1);
#if USE_TDM
    v16h bf = *(const v16h*)(bstage + ((size_t)kt * 32 + lane) * 16);
#else
    v16h bf = *(const v16h*)(b + ((size_t)kt * 32 + lane) * 16);
#endif
    acc = wmma_f16(af, bf, acc);
  }
  int hi = lane >> 4, col = lane & 15;
#pragma unroll
  for (int r = 0; r < 8; ++r) {
    int mloc = r + hi * 8;
    float v = acc[r] * alpha;
    int ncol = nt * 16 + col;
    if (biasN) v += biasN[ncol];
    if (actGelu) v = gelu_tanh(v);
    size_t idx = (size_t)(mt * 16 + mloc) * N + ncol;
    if (resid) v += resid[idx];
    if (Cout) Cout[idx] = v;
    if (apackOut) store_apack(apackOut, mt, KTdst, mloc, ncol, v);
  }
}

// ------------------------------- attention ---------------------------------
// One wave per (head, 16-query tile); flash-style online softmax over 32-key
// chunks; P re-shaped to A-fragment layout through per-wave LDS.
__global__ void attn_kernel(const _Float16* __restrict__ qP,  // [H][32][2][32][16]
                            const _Float16* __restrict__ kP,  // [H][32][2][32][16]
                            const _Float16* __restrict__ vP,  // [H][4][32][32][16]
                            const int* __restrict__ amask, float* __restrict__ ao) {
  __shared__ _Float16 pT[4][16 * 32];
  int lane = threadIdx.x & 31;
  int w = threadIdx.x >> 5;
  int lin = blockIdx.x * 4 + w;
  int head = lin >> 5, mt = lin & 31;
  int hi = lane >> 4, col = lane & 15;

  const _Float16* qBase = qP + (size_t)head * 32768 + (size_t)mt * 1024;
  const _Float16* kHead = kP + (size_t)head * 32768;
  const _Float16* vHead = vP + (size_t)head * 65536;

  v8f o0 = {}, o1 = {}, o2 = {}, o3 = {};
  float rMax[8], rSum[8];
#pragma unroll
  for (int r = 0; r < 8; ++r) { rMax[r] = -1e30f; rSum[r] = 0.0f; }

  v16h qf0 = *(const v16h*)(qBase + ((size_t)0 * 32 + lane) * 16);
  v16h qf1 = *(const v16h*)(qBase + ((size_t)1 * 32 + lane) * 16);

  for (int kc = 0; kc < 16; ++kc) {          // 32-key chunk
    float pv[2][8];
    float cmax[8];
#pragma unroll
    for (int r = 0; r < 8; ++r) cmax[r] = -1e30f;
#pragma unroll
    for (int t = 0; t < 2; ++t) {            // two 16-key score tiles
      int nt = kc * 2 + t;
      const _Float16* kBase = kHead + (size_t)nt * 1024;
      v16h kf0 = *(const v16h*)(kBase + ((size_t)0 * 32 + lane) * 16);
      v16h kf1 = *(const v16h*)(kBase + ((size_t)1 * 32 + lane) * 16);
      v8f s = {};
      s = wmma_f16(qf0, kf0, s);
      s = wmma_f16(qf1, kf1, s);
      int colg = nt * 16 + col;
      float am = (float)amask[colg];
#pragma unroll
      for (int r = 0; r < 8; ++r) {
        int rowg = mt * 16 + r + hi * 8;
        float cm = (colg <= rowg) ? am : 0.0f;
        if (colg == rowg) cm = 1.0f;          // diagonal forced visible
        float sv = s[r] * 0.125f - (1.0f - cm) * 1e9f;
        pv[t][r] = sv;
        cmax[r] = fmaxf(cmax[r], sv);
      }
    }
    // online softmax update per row (rows live on 16-lane halves)
#pragma unroll
    for (int r = 0; r < 8; ++r) {
      float m = hmax16(cmax[r]);
      float nm = fmaxf(rMax[r], m);
      float f = __expf(rMax[r] - nm);
      rMax[r] = nm;
      float p0 = __expf(pv[0][r] - nm);
      float p1 = __expf(pv[1][r] - nm);
      rSum[r] = rSum[r] * f + hsum16(p0 + p1);
      o0[r] *= f; o1[r] *= f; o2[r] *= f; o3[r] *= f;
      int mloc = r + hi * 8;
      pT[w][mloc * 32 + col]      = (_Float16)p0;
      pT[w][mloc * 32 + 16 + col] = (_Float16)p1;
    }
    // reshape P (C-layout) -> A fragment via LDS (same-wave DS is in-order)
    v16h pa;
#pragma unroll
    for (int h = 0; h < 16; ++h)
      pa[h] = pT[w][(lane & 15) * 32 + afrag_k(lane, h)];
    v16h v0 = *(const v16h*)(vHead + (((size_t)0 * 32 + kc) * 32 + lane) * 16);
    v16h v1 = *(const v16h*)(vHead + (((size_t)1 * 32 + kc) * 32 + lane) * 16);
    v16h v2 = *(const v16h*)(vHead + (((size_t)2 * 32 + kc) * 32 + lane) * 16);
    v16h v3 = *(const v16h*)(vHead + (((size_t)3 * 32 + kc) * 32 + lane) * 16);
    o0 = wmma_f16(pa, v0, o0);
    o1 = wmma_f16(pa, v1, o1);
    o2 = wmma_f16(pa, v2, o2);
    o3 = wmma_f16(pa, v3, o3);
  }
#pragma unroll
  for (int r = 0; r < 8; ++r) {
    int q = mt * 16 + r + hi * 8;
    float inv = 1.0f / rSum[r];
    size_t base = (size_t)q * cD + head * cHD + col;
    ao[base + 0]  = o0[r] * inv;
    ao[base + 16] = o1[r] * inv;
    ao[base + 32] = o2[r] * inv;
    ao[base + 48] = o3[r] * inv;
  }
}

// ------------------------------- MoE routing -------------------------------
__global__ void router_kernel(const float* __restrict__ logits, float* __restrict__ gateVal,
                              int* __restrict__ expCnt, int* __restrict__ expList) {
  int t = blockIdx.x * blockDim.x + threadIdx.x;
  if (t >= cS) return;
  float l[cE], mx = -1e30f;
#pragma unroll
  for (int e = 0; e < cE; ++e) { l[e] = logits[t * cE + e]; mx = fmaxf(mx, l[e]); }
  float sum = 0.0f;
#pragma unroll
  for (int e = 0; e < cE; ++e) { l[e] = __expf(l[e] - mx); sum += l[e]; }
  float inv = 1.0f / sum;
  int i1 = 0; float v1 = -1.0f;
#pragma unroll
  for (int e = 0; e < cE; ++e) { float p = l[e] * inv; if (p > v1) { v1 = p; i1 = e; } }
  int i2 = -1; float v2 = -1.0f;
#pragma unroll
  for (int e = 0; e < cE; ++e) {
    if (e == i1) continue;
    float p = l[e] * inv;
    if (p > v2) { v2 = p; i2 = e; }
  }
  gateVal[t * cTK + 0] = v1;
  gateVal[t * cTK + 1] = v2;
  int p0 = atomicAdd(&expCnt[i1], 1);
  expList[i1 * cS + p0] = t * cTK + 0;
  int p1 = atomicAdd(&expCnt[i2], 1);
  expList[i2 * cS + p1] = t * cTK + 1;
}

// single-thread scan: pad each expert's assignments to 16-row tiles
__global__ void moe_scan_kernel(const int* __restrict__ expCnt, const int* __restrict__ expList,
                                int* __restrict__ tileExpert, int* __restrict__ rowAssign) {
  if (threadIdx.x != 0 || blockIdx.x != 0) return;
  int tile = 0;
  for (int e = 0; e < cE; ++e) {
    int c = expCnt[e];
    int tiles = (c + 15) >> 4;
    for (int tt = 0; tt < tiles && tile < MAX_TILES; ++tt, ++tile) {
      tileExpert[tile] = e;
      for (int i = 0; i < 16; ++i) {
        int j = tt * 16 + i;
        rowAssign[tile * 16 + i] = (j < c) ? expList[e * cS + j] : -1;
      }
    }
  }
  for (; tile < MAX_TILES; ++tile) {
    tileExpert[tile] = -1;
    for (int i = 0; i < 16; ++i) rowAssign[tile * 16 + i] = -1;
  }
}

// gather h2 rows for assignments into A-fragment layout (zeros for padding)
__global__ void moe_gather_kernel(const float* __restrict__ h2, const int* __restrict__ rowAssign,
                                  _Float16* __restrict__ gatherPack) {
  int frag = blockIdx.x;                 // tt*16 + kt, KT = 16
  int tt = frag >> 4, kt = frag & 15;
  int lane = threadIdx.x;
  int as = rowAssign[tt * 16 + (lane & 15)];
  _Float16* d = gatherPack + ((size_t)frag * 32 + lane) * 16;
  if (as < 0) {
#pragma unroll
    for (int h = 0; h < 16; ++h) d[h] = (_Float16)0.0f;
    return;
  }
  int token = as >> 1;
#pragma unroll
  for (int h = 0; h < 16; ++h)
    d[h] = (_Float16)h2[(size_t)token * cD + kt * 32 + afrag_k(lane, h)];
}

// grouped GEMM1: hid = gelu(gathered_h2 @ ek[e]); store in A-frag layout
__global__ void moe_gemm1_kernel(const _Float16* __restrict__ gatherPack,
                                 const _Float16* __restrict__ ekPack,
                                 const int* __restrict__ tileExpert,
                                 _Float16* __restrict__ hidPack) {
  int tt = blockIdx.x, nt = blockIdx.y, lane = threadIdx.x;
  int e = tileExpert[tt];
  if (e < 0) return;
  const _Float16* a = gatherPack + (size_t)tt * 16 * 512;       // KT=16
  const _Float16* b = ekPack + ((size_t)e * 32 + nt) * 16 * 512;
  v8f acc = {};
  for (int kt = 0; kt < 16; ++kt) {
    v16h af = *(const v16h*)(a + ((size_t)kt * 32 + lane) * 16);
    v16h bf = *(const v16h*)(b + ((size_t)kt * 32 + lane) * 16);
    acc = wmma_f16(af, bf, acc);
  }
  int hi = lane >> 4, col = lane & 15;
#pragma unroll
  for (int r = 0; r < 8; ++r)
    store_apack(hidPack, tt, 16, r + hi * 8, nt * 16 + col, gelu_tanh(acc[r]));
}

// grouped GEMM2: moe[token] += gate * (hid @ ev[e]) via global f32 atomics
__global__ void moe_gemm2_kernel(const _Float16* __restrict__ hidPack,
                                 const _Float16* __restrict__ evPack,
                                 const int* __restrict__ tileExpert,
                                 const int* __restrict__ rowAssign,
                                 const float* __restrict__ gateVal,
                                 float* __restrict__ moeAcc) {
  int tt = blockIdx.x, nt = blockIdx.y, lane = threadIdx.x;
  int e = tileExpert[tt];
  if (e < 0) return;
  const _Float16* a = hidPack + (size_t)tt * 16 * 512;          // KT=16 (FF=512)
  const _Float16* b = evPack + ((size_t)e * 32 + nt) * 16 * 512;
  v8f acc = {};
  for (int kt = 0; kt < 16; ++kt) {
    v16h af = *(const v16h*)(a + ((size_t)kt * 32 + lane) * 16);
    v16h bf = *(const v16h*)(b + ((size_t)kt * 32 + lane) * 16);
    acc = wmma_f16(af, bf, acc);
  }
  int hi = lane >> 4, col = lane & 15;
#pragma unroll
  for (int r = 0; r < 8; ++r) {
    int as = rowAssign[tt * 16 + r + hi * 8];
    if (as < 0) continue;
    int token = as >> 1;
    atomicAdd(&moeAcc[(size_t)token * cD + nt * 16 + col], acc[r] * gateVal[as]);
  }
}

// ------------------------------- launcher ----------------------------------
extern "C" void kernel_launch(void* const* d_in, const int* in_sizes, int n_in,
                              void* d_out, int out_size, void* d_ws, size_t ws_size,
                              hipStream_t stream) {
  (void)in_sizes; (void)n_in; (void)out_size; (void)ws_size;

  const float* x      = (const float*)d_in[0];
  const int*   amask  = (const int*)d_in[1];
  const float* wq     = (const float*)d_in[2];
  const float* wk     = (const float*)d_in[3];
  const float* wv     = (const float*)d_in[4];
  const float* wo     = (const float*)d_in[5];
  const float* a1     = (const float*)d_in[6];
  const float* g1     = (const float*)d_in[7];
  const float* b1     = (const float*)d_in[8];
  const float* a2     = (const float*)d_in[9];
  const float* g2     = (const float*)d_in[10];
  const float* b2     = (const float*)d_in[11];
  const float* gate_w = (const float*)d_in[12];
  const float* gate_b = (const float*)d_in[13];
  const float* ek     = (const float*)d_in[14];
  const float* ev     = (const float*)d_in[15];
  const float* sk     = (const float*)d_in[16];
  const float* sv     = (const float*)d_in[17];
  float* out = (float*)d_out;

  // ----- workspace bump allocator -----
  char* p = (char*)d_ws;
  auto alloc = [&p](size_t bytes) -> char* {
    char* r = p;
    p += (bytes + 255) & ~(size_t)255;
    return r;
  };
  const size_t SD = (size_t)cS * cD;                 // 262144
  float*     h1       = (float*)alloc(SD * 4);
  _Float16*  h1Pack   = (_Float16*)alloc(SD * 2);
  _Float16*  wqPack   = (_Float16*)alloc(SD * 2);
  _Float16*  wkPack   = (_Float16*)alloc(SD * 2);
  _Float16*  wvPack   = (_Float16*)alloc(SD * 2);
  _Float16*  woPack   = (_Float16*)alloc(SD * 2);
  _Float16*  gwPack   = (_Float16*)alloc((size_t)cD * cE * 2);
  _Float16*  ekPack   = (_Float16*)alloc((size_t)cE * cD * cFF * 2);
  _Float16*  evPack   = (_Float16*)alloc((size_t)cE * cFF * cD * 2);
  _Float16*  skPack   = (_Float16*)alloc((size_t)cD * cFF * 2);
  _Float16*  svPack   = (_Float16*)alloc((size_t)cFF * cD * 2);
  float*     qf       = (float*)alloc(SD * 4);
  float*     kf       = (float*)alloc(SD * 4);
  float*     vf       = (float*)alloc(SD * 4);
  _Float16*  qPack    = (_Float16*)alloc(SD * 2);
  _Float16*  kPack    = (_Float16*)alloc(SD * 2);
  _Float16*  vPack    = (_Float16*)alloc(SD * 2);
  float*     ao       = (float*)alloc(SD * 4);
  _Float16*  aoPack   = (_Float16*)alloc(SD * 2);
  float*     x1       = (float*)alloc(SD * 4);
  float*     h2       = (float*)alloc(SD * 4);
  _Float16*  h2Pack   = (_Float16*)alloc(SD * 2);
  float*     logits   = (float*)alloc((size_t)cS * cE * 4);
  float*     gateVal  = (float*)alloc((size_t)cS * cTK * 4);
  int*       expCnt   = (int*)alloc(cE * 4);
  int*       expList  = (int*)alloc((size_t)cE * cS * 4);
  int*       tileExpert = (int*)alloc(MAX_TILES * 4);
  int*       rowAssign  = (int*)alloc(MAX_TILES * 16 * 4);
  _Float16*  gatherPack = (_Float16*)alloc((size_t)MAX_TILES * 16 * cD * 2);
  _Float16*  hidPack    = (_Float16*)alloc((size_t)MAX_TILES * 16 * cFF * 2);
  _Float16*  hsPack     = (_Float16*)alloc((size_t)cS * cFF * 2);
  float*     moeAcc     = (float*)alloc(SD * 4);
  float*     shAcc      = (float*)alloc(SD * 4);

  // ----- zero accumulators / counters -----
  zero_kernel<<<1024, 256, 0, stream>>>(moeAcc, (int)SD);
  zero_kernel<<<1, 32, 0, stream>>>((float*)expCnt, cE);

  // ----- DyT pre-norm #1 + pack -----
  dyt_kernel<<<1024, 256, 0, stream>>>(x, a1, g1, b1, h1, (int)SD);
  pack_a<<<dim3(512, 1, 1), 32, 0, stream>>>(h1, h1Pack, 16, cD, 0, 0, 0, 0);

  // ----- pack weights (one-time f32 -> f16 fragment layout) -----
  pack_b<<<dim3(512, 1, 1), 32, 0, stream>>>(wq, wqPack, 16, cD, 1, 0, 0, 0, 0);
  pack_b<<<dim3(512, 1, 1), 32, 0, stream>>>(wk, wkPack, 16, cD, 1, 0, 0, 0, 0);
  pack_b<<<dim3(512, 1, 1), 32, 0, stream>>>(wv, wvPack, 16, cD, 1, 0, 0, 0, 0);
  pack_b<<<dim3(512, 1, 1), 32, 0, stream>>>(wo, woPack, 16, cD, 1, 0, 0, 0, 0);
  pack_b<<<dim3(16, 1, 1),  32, 0, stream>>>(gate_w, gwPack, 16, cE, 1, 0, 0, 0, 0);
  pack_b<<<dim3(512, 1, cE), 32, 0, stream>>>(ek, ekPack, 16, cFF, 1, 0, 0,
                                              (size_t)cD * cFF, (size_t)cD * cFF);
  pack_b<<<dim3(512, 1, cE), 32, 0, stream>>>(ev, evPack, 16, cD, 1, 0, 0,
                                              (size_t)cFF * cD, (size_t)cFF * cD);
  pack_b<<<dim3(512, 1, 1), 32, 0, stream>>>(sk, skPack, 16, cFF, 1, 0, 0, 0, 0);
  pack_b<<<dim3(512, 1, 1), 32, 0, stream>>>(sv, svPack, 16, cD, 1, 0, 0, 0, 0);

  // ----- QKV projections (WMMA GEMMs, TDM-staged B) -----
  gemm16<<<dim3(8, 32), 128, 0, stream>>>(h1Pack, wqPack, qf, nullptr, nullptr, nullptr,
                                          16, cD, 0, 0, 1.0f);
  gemm16<<<dim3(8, 32), 128, 0, stream>>>(h1Pack, wkPack, kf, nullptr, nullptr, nullptr,
                                          16, cD, 0, 0, 1.0f);
  gemm16<<<dim3(8, 32), 128, 0, stream>>>(h1Pack, wvPack, vf, nullptr, nullptr, nullptr,
                                          16, cD, 0, 0, 1.0f);

  // ----- RoPE on Q,K then per-head fragment packs -----
  rope_kernel<<<512, 256, 0, stream>>>(qf, kf);
  // Q as A (per head, K-dim = HD=64 -> KT=2)
  pack_a<<<dim3(64, 1, cH), 32, 0, stream>>>(qf, qPack, 2, cD, 0, cHD, 0, 32768);
  // K^T as B for scores: (k=hd, n=key) -> kf[key*D + head*64 + hd]
  pack_b<<<dim3(64, 1, cH), 32, 0, stream>>>(kf, kPack, 2, 1, cD, 0, cHD, 0, 32768);
  // V as B for P*V: (k=key, n=hd) -> vf[key*D + head*64 + hd]
  pack_b<<<dim3(128, 1, cH), 32, 0, stream>>>(vf, vPack, 32, cD, 1, 0, cHD, 0, 65536);

  // ----- flash attention -----
  attn_kernel<<<64, 128, 0, stream>>>(qPack, kPack, vPack, amask, ao);

  // ----- output projection + residual -----
  pack_a<<<dim3(512, 1, 1), 32, 0, stream>>>(ao, aoPack, 16, cD, 0, 0, 0, 0);
  gemm16<<<dim3(8, 32), 128, 0, stream>>>(aoPack, woPack, x1, x, nullptr, nullptr,
                                          16, cD, 0, 0, 1.0f);

  // ----- DyT pre-norm #2 + pack -----
  dyt_kernel<<<1024, 256, 0, stream>>>(x1, a2, g2, b2, h2, (int)SD);
  pack_a<<<dim3(512, 1, 1), 32, 0, stream>>>(h2, h2Pack, 16, cD, 0, 0, 0, 0);

  // ----- router: logits GEMM (N=16) + softmax/top-2 + expert lists -----
  gemm16<<<dim3(8, 1), 128, 0, stream>>>(h2Pack, gwPack, logits, nullptr, gate_b, nullptr,
                                         16, cE, 0, 0, 1.0f);
  router_kernel<<<2, 256, 0, stream>>>(logits, gateVal, expCnt, expList);
  moe_scan_kernel<<<1, 32, 0, stream>>>(expCnt, expList, tileExpert, rowAssign);

  // ----- grouped MoE (expert weights stream exactly once through WMMA) -----
  moe_gather_kernel<<<MAX_TILES * 16, 32, 0, stream>>>(h2, rowAssign, gatherPack);
  moe_gemm1_kernel<<<dim3(MAX_TILES, 32), 32, 0, stream>>>(gatherPack, ekPack, tileExpert,
                                                           hidPack);
  moe_gemm2_kernel<<<dim3(MAX_TILES, 32), 32, 0, stream>>>(hidPack, evPack, tileExpert,
                                                           rowAssign, gateVal, moeAcc);

  // ----- shared expert (dense): hs = gelu(h2 @ sk); sh = hs @ sv -----
  gemm16<<<dim3(8, 32), 128, 0, stream>>>(h2Pack, skPack, nullptr, nullptr, nullptr, hsPack,
                                          16, cFF, 16, 1, 1.0f);
  gemm16<<<dim3(8, 32), 128, 0, stream>>>(hsPack, svPack, shAcc, nullptr, nullptr, nullptr,
                                          16, cD, 0, 0, 1.0f);

  // ----- out = x1 + moe + shared -----
  final_add_kernel<<<1024, 256, 0, stream>>>(x1, moeAcc, shAcc, out, (int)SD);
}
